// ArcFaceClassifier_6622839570773
// MI455X (gfx1250) — matrix operand order repaired and checked by reading, not compiled, not run on hardware
//
#include <hip/hip_runtime.h>

// ===========================================================================
// ArcFace forward for MI455X (gfx1250, wave32, WMMA).
//
//   cos  = (x@U) * (1/||x||)             U = W / ||W[:,j]||
//   dist = x_sq + (s1*||u||)^2 - 2*s1*(x@U)
//   arc  = -(Rn @ U),  R = x - (s1[t]-4)*u_t
// => ONE fused bf16 WMMA GEMM [x ; Rn](8192x512) @ U(512x10575), f32 accum.
// Roofline: 550 MB output writes -> ~24us floor @ 23.3 TB/s. bf16 WMMA
// (16x16x32) keeps 88.7 GFLOP of matrix work at that floor (f32 WMMA K=4
// would be ~8x slower). B operand (10.8 MB) is L2-resident; outputs are
// streamed with non-temporal stores so they don't evict it.
// Staging: double-buffered GLOBAL_LOAD_ASYNC_TO_LDS_B128 (ASYNCcnt-tracked,
// no VGPR round-trip); signature fixed to (as1 v4i*, as3 v4i*, imm, imm)
// per the round-2 diagnostic.
// ===========================================================================

typedef __bf16 bf16;
typedef __attribute__((ext_vector_type(16))) __bf16 v16bf;
typedef __attribute__((ext_vector_type(8)))  __bf16 v8bf;
typedef __attribute__((ext_vector_type(8)))  float  v8f;

#define NCLS 10575
#define KDIM 512
#define BS   4096

// ---- workspace layout (bytes); total ~19.6 MB ----
#define WS_AB    0ull            // bf16 [8192][512]  rows 0..4095 = x, 4096.. = Rn
#define WS_UT    8388608ull      // bf16 [10575][512] row j = unit column j of W
#define WS_INVN  19217408ull     // f32 [10575]
#define WS_U2    19386624ull     // f32 [10575]
#define WS_XSQ   19555840ull     // f32 [4096]
#define WS_INVX  19572224ull     // f32 [4096]

// ---- output offsets (f32 elements), concatenated in return order ----
#define OFF_COS  0ull
#define OFF_ARC  43315200ull
#define OFF_DIST 86630400ull
#define OFF_DISN 129945600ull
#define OFF_X    129949696ull
#define OFF_W    132046848ull

// ---- CDNA5 async global->LDS path (probe via __has_builtin) ----
#if defined(__gfx1250__) &&                                          \
    __has_builtin(__builtin_amdgcn_global_load_async_to_lds_b128) && \
    __has_builtin(__builtin_amdgcn_s_wait_asynccnt)
#define USE_ASYNC_LDS 1
#endif

#ifdef USE_ASYNC_LDS
typedef int vi4 __attribute__((vector_size(16)));
typedef __attribute__((address_space(1))) vi4 g_vi4;  // global int4
typedef __attribute__((address_space(3))) vi4 l_vi4;  // LDS int4 (32-bit ptr)

__device__ __forceinline__ void async_copy_b128(const void* g, void* l) {
  // AS3 pointers are 32-bit; low 32 bits of a generic shared pointer are the
  // LDS byte offset (ISA flat-aperture mapping), so truncation is exact.
  __builtin_amdgcn_global_load_async_to_lds_b128(
      (g_vi4*)(size_t)g, (l_vi4*)(unsigned int)(size_t)l, 0, 0);
}
#endif

__device__ __forceinline__ float waveReduce32(float v) {
#pragma unroll
  for (int off = 16; off; off >>= 1) v += __shfl_xor(v, off, 32);
  return v;
}

// ---------------------------------------------------------------------------
// K0: per-column norms of W, write bf16 U^T (row-contiguous unit columns).
// ---------------------------------------------------------------------------
__global__ __launch_bounds__(256) void col_prep(
    const float* __restrict__ W, bf16* __restrict__ Ut,
    float* __restrict__ invn, float* __restrict__ u2) {
  __shared__ float part[8][32];
  __shared__ float invLds[32];
  const int tx = threadIdx.x & 31, ty = threadIdx.x >> 5;
  const int j = blockIdx.x * 32 + tx;
  float s = 0.f;
  if (j < NCLS) {
#pragma unroll 4
    for (int k = ty; k < KDIM; k += 8) {
      float v = W[(size_t)k * NCLS + j];
      s = fmaf(v, v, s);
    }
  }
  part[ty][tx] = s;
  __syncthreads();
  if (ty == 0 && j < NCLS) {
    float tot = 0.f;
#pragma unroll
    for (int q = 0; q < 8; ++q) tot += part[q][tx];
    float inv = 1.0f / fmaxf(sqrtf(tot), 1e-12f);
    invn[j] = inv;
    u2[j] = tot * inv * inv;
    invLds[tx] = inv;
  }
  __syncthreads();
  if (j < NCLS) {
    float inv = invLds[tx];
#pragma unroll 4
    for (int k = ty; k < KDIM; k += 8)
      Ut[(size_t)j * KDIM + k] = (bf16)(W[(size_t)k * NCLS + j] * inv);
  }
}

// ---------------------------------------------------------------------------
// K1: x row norms + bf16 cast into A buffer (rows 0..4095). 1 wave per row.
// ---------------------------------------------------------------------------
__global__ __launch_bounds__(256) void row_prep(
    const float* __restrict__ x, bf16* __restrict__ Xb,
    float* __restrict__ xsq, float* __restrict__ invx) {
  const int wv = threadIdx.x >> 5, lane = threadIdx.x & 31;
  const int i = blockIdx.x * 8 + wv;
  const float4* xr = (const float4*)(x + (size_t)i * KDIM);
  float4 v[4];
  float s = 0.f;
#pragma unroll
  for (int t = 0; t < 4; ++t) {
    v[t] = xr[lane + 32 * t];
    s += v[t].x * v[t].x + v[t].y * v[t].y + v[t].z * v[t].z + v[t].w * v[t].w;
  }
  s = waveReduce32(s);
  if (lane == 0) {
    xsq[i] = s;
    invx[i] = 1.0f / fmaxf(sqrtf(s), 1e-12f);
  }
#pragma unroll
  for (int t = 0; t < 4; ++t) {
    bf16* d = Xb + (size_t)i * KDIM + (size_t)(lane + 32 * t) * 4;
    d[0] = (bf16)v[t].x; d[1] = (bf16)v[t].y;
    d[2] = (bf16)v[t].z; d[3] = (bf16)v[t].w;
  }
}

// ---------------------------------------------------------------------------
// K2: R = x - (s1[t]-4)*u_t from CONTIGUOUS Ut rows; normalize; bf16 into
// A rows 4096..8191. Also dis_norm[i].
// ---------------------------------------------------------------------------
__global__ __launch_bounds__(256) void r_prep(
    const float* __restrict__ x, const int* __restrict__ targ,
    const float* __restrict__ s1, const bf16* __restrict__ Ut,
    const float* __restrict__ u2, const float* __restrict__ xsq,
    bf16* __restrict__ Rnb, float* __restrict__ disn) {
  const int wv = threadIdx.x >> 5, lane = threadIdx.x & 31;
  const int i = blockIdx.x * 8 + wv;
  const int tg = targ[i];
  const float coef = s1[tg] - 4.0f;
  const bf16* u = Ut + (size_t)tg * KDIM;
  const float* xr = x + (size_t)i * KDIM;
  float r[16];
  float s = 0.f;
#pragma unroll
  for (int t = 0; t < 16; ++t) {
    int k = lane + 32 * t;
    float rv = xr[k] - coef * (float)u[k];
    r[t] = rv;
    s = fmaf(rv, rv, s);
  }
  s = waveReduce32(s);
  float rinv = 1.0f / fmaxf(sqrtf(s), 1e-12f);
  bf16* d = Rnb + (size_t)i * KDIM;
#pragma unroll
  for (int t = 0; t < 16; ++t) d[lane + 32 * t] = (bf16)(r[t] * rinv);
  if (lane == 0) {
    float dd = s1[tg] * sqrtf(u2[tg]) - sqrtf(xsq[i]);
    disn[i] = dd * dd;
  }
}

// ---------------------------------------------------------------------------
// Main fused GEMM: C(8192x10575) = A @ Ut^T via bf16 WMMA, f32 accumulate.
// 128x128 block tile, 4 waves x (64x64) = 4x4 v8f accumulators each.
// LDS rows padded to stride 40 (80 B -> 20-bank step, conflict-free frags).
// Async double-buffered staging when GLOBAL_LOAD_ASYNC_TO_LDS is available.
// ---------------------------------------------------------------------------
__global__ __launch_bounds__(128) void gemm_epilogue(
    const bf16* __restrict__ Ab, const bf16* __restrict__ Ut,
    const float* __restrict__ s1, const float* __restrict__ u2,
    const float* __restrict__ xsq, const float* __restrict__ invx,
    float* __restrict__ out) {
#ifdef USE_ASYNC_LDS
  __shared__ bf16 As[2][128 * 40];
  __shared__ bf16 Bs[2][128 * 40];
#else
  __shared__ bf16 As[1][128 * 40];
  __shared__ bf16 Bs[1][128 * 40];
#endif
  const int tid = threadIdx.x;
  const int wv = tid >> 5, lane = tid & 31;
  const int lo = lane & 15, hi = lane >> 4;
  const int wm = wv >> 1, wn = wv & 1;
  const int m0 = blockIdx.y * 128;
  const int n0 = blockIdx.x * 128;

  v8f acc[4][4];
#pragma unroll
  for (int mt = 0; mt < 4; ++mt)
#pragma unroll
    for (int nt = 0; nt < 4; ++nt) {
      v8f z = {0.f, 0.f, 0.f, 0.f, 0.f, 0.f, 0.f, 0.f};
      acc[mt][nt] = z;
    }

#ifdef USE_ASYNC_LDS
  // ---- issue helper: 8 async b128 per wave per tile (4 A + 4 B) ----
  auto issue_tile = [&](int buf, int kk) {
#pragma unroll
    for (int it = 0; it < 4; ++it) {
      int u = tid + it * 128;
      int row = u >> 2, ch = u & 3;
      async_copy_b128(&Ab[(size_t)(m0 + row) * KDIM + kk + ch * 8],
                      &As[buf][row * 40 + ch * 8]);
      int gj = n0 + row;
      if (gj > NCLS - 1) gj = NCLS - 1;  // clamp; guarded at store
      async_copy_b128(&Ut[(size_t)gj * KDIM + kk + ch * 8],
                      &Bs[buf][row * 40 + ch * 8]);
    }
  };
  issue_tile(0, 0);
  int cur = 0;
#endif

  for (int kk = 0; kk < KDIM; kk += 32) {
#ifdef USE_ASYNC_LDS
    if (kk + 32 < KDIM) {
      issue_tile(cur ^ 1, kk + 32);          // prefetch next tile into other buf
      __builtin_amdgcn_s_wait_asynccnt(8);   // current tile's 8 ops complete
    } else {
      __builtin_amdgcn_s_wait_asynccnt(0);
    }
    __syncthreads();                         // all waves' async groups visible
    const bf16* AsT = As[cur];
    const bf16* BsT = Bs[cur];
#else
#pragma unroll
    for (int it = 0; it < 4; ++it) {
      int u = tid + it * 128;
      int row = u >> 2, ch = u & 3;
      *(v8bf*)&As[0][row * 40 + ch * 8] =
          *(const v8bf*)&Ab[(size_t)(m0 + row) * KDIM + kk + ch * 8];
      int gj = n0 + row;
      if (gj > NCLS - 1) gj = NCLS - 1;
      *(v8bf*)&Bs[0][row * 40 + ch * 8] =
          *(const v8bf*)&Ut[(size_t)gj * KDIM + kk + ch * 8];
    }
    if (kk + 32 < KDIM) {  // gfx1250 global_prefetch_b8 for next K chunk
#pragma unroll
      for (int it = 0; it < 4; ++it) {
        int u = tid + it * 128;
        int row = u >> 2, ch = u & 3;
        __builtin_prefetch(&Ab[(size_t)(m0 + row) * KDIM + kk + 32 + ch * 8], 0, 1);
        int gj = n0 + row;
        if (gj > NCLS - 1) gj = NCLS - 1;
        __builtin_prefetch(&Ut[(size_t)gj * KDIM + kk + 32 + ch * 8], 0, 1);
      }
    }
    __syncthreads();
    const bf16* AsT = As[0];
    const bf16* BsT = Bs[0];
#endif

    // A 16-bit 16x32 frag: lanes<16 hold K {0..7,16..23}, lanes>=16 {8..15,24..31}
    // B 16-bit 32x16 frag: lane holds N=lo, K {0..7,16..23} (lanes<16) / {8..15,24..31} (lanes>=16)
    v16bf a[4], b[4];
#pragma unroll
    for (int mt = 0; mt < 4; ++mt) {
      const bf16* p = &AsT[(wm * 64 + mt * 16 + lo) * 40 + hi * 8];
      v8bf x0 = *(const v8bf*)p;
      v8bf x1 = *(const v8bf*)(p + 16);
#pragma unroll
      for (int e = 0; e < 8; ++e) { a[mt][e] = x0[e]; a[mt][e + 8] = x1[e]; }
    }
#pragma unroll
    for (int nt = 0; nt < 4; ++nt) {
      const bf16* p = &BsT[(wn * 64 + nt * 16 + lo) * 40 + hi * 8];
      v8bf x0 = *(const v8bf*)p;
      v8bf x1 = *(const v8bf*)(p + 16);
#pragma unroll
      for (int e = 0; e < 8; ++e) { b[nt][e] = x0[e]; b[nt][e + 8] = x1[e]; }
    }
#pragma unroll
    for (int mt = 0; mt < 4; ++mt)
#pragma unroll
      for (int nt = 0; nt < 4; ++nt)
        acc[mt][nt] = __builtin_amdgcn_wmma_f32_16x16x32_bf16(
            false, a[mt], false, b[nt], (short)0, acc[mt][nt], false, false);
    __syncthreads();   // reads done before next tile overwrites the other buf
#ifdef USE_ASYNC_LDS
    cur ^= 1;
#endif
  }

  // ---- fused epilogue: streaming non-temporal stores (don't evict L2 B) ----
  float* cosL = out + OFF_COS;
  float* arcL = out + OFF_ARC;
  float* dist = out + OFF_DIST;
  const bool isX = (m0 < BS);  // 128-row tiles never straddle the x/Rn halves
#pragma unroll
  for (int mt = 0; mt < 4; ++mt) {
    const int rbase = m0 + wm * 64 + mt * 16 + hi * 8;  // C: M = hi*8 + r
    float xs8[8], iv8[8];
    if (isX) {
#pragma unroll
      for (int r = 0; r < 8; ++r) { xs8[r] = xsq[rbase + r]; iv8[r] = invx[rbase + r]; }
    }
#pragma unroll
    for (int nt = 0; nt < 4; ++nt) {
      const int j = n0 + wn * 64 + nt * 16 + lo;  // C: N = lo
      if (j >= NCLS) continue;
      if (isX) {
        float sj = s1[j];
        float csq = sj * sj * u2[j];
#pragma unroll
        for (int r = 0; r < 8; ++r) {
          float g = acc[mt][nt][r];
          size_t idx = (size_t)(rbase + r) * NCLS + j;
          __builtin_nontemporal_store(g * iv8[r], cosL + idx);
          __builtin_nontemporal_store(fmaf(-2.0f * sj, g, xs8[r] + csq), dist + idx);
        }
      } else {
#pragma unroll
        for (int r = 0; r < 8; ++r) {
          size_t idx = (size_t)(rbase - BS + r) * NCLS + j;
          __builtin_nontemporal_store(-acc[mt][nt][r], arcL + idx);
        }
      }
    }
  }
}

// ---------------------------------------------------------------------------
extern "C" void kernel_launch(void* const* d_in, const int* in_sizes, int n_in,
                              void* d_out, int out_size, void* d_ws, size_t ws_size,
                              hipStream_t stream) {
  (void)in_sizes; (void)n_in; (void)out_size; (void)ws_size;
  const float* x  = (const float*)d_in[0];
  const int*   tg = (const int*)d_in[1];   // targ (class indices)
  const float* s1 = (const float*)d_in[2];
  const float* W  = (const float*)d_in[3];
  float* out = (float*)d_out;
  char* ws = (char*)d_ws;

  bf16*  Ab   = (bf16*)(ws + WS_AB);
  bf16*  Ut   = (bf16*)(ws + WS_UT);
  float* invn = (float*)(ws + WS_INVN);
  float* u2   = (float*)(ws + WS_U2);
  float* xsq  = (float*)(ws + WS_XSQ);
  float* invx = (float*)(ws + WS_INVX);

  col_prep<<<dim3((NCLS + 31) / 32), 256, 0, stream>>>(W, Ut, invn, u2);
  row_prep<<<dim3(BS / 8), 256, 0, stream>>>(x, Ab, xsq, invx);
  r_prep<<<dim3(BS / 8), 256, 0, stream>>>(x, tg, s1, Ut, u2, xsq,
                                           Ab + (size_t)BS * KDIM, out + OFF_DISN);
  gemm_epilogue<<<dim3((NCLS + 127) / 128, (2 * BS) / 128), 128, 0, stream>>>(
      Ab, Ut, s1, u2, xsq, invx, out);

  // pass-through outputs
  (void)hipMemcpyAsync(out + OFF_X, x, (size_t)BS * KDIM * sizeof(float),
                       hipMemcpyDeviceToDevice, stream);
  (void)hipMemcpyAsync(out + OFF_W, W, (size_t)KDIM * NCLS * sizeof(float),
                       hipMemcpyDeviceToDevice, stream);
}
